// AGCN_20684562498263
// MI455X (gfx1250) — matrix operand order repaired and checked by reading, not compiled, not run on hardware
//
#include <hip/hip_runtime.h>
#include <hip/hip_bf16.h>
#include <math.h>

// ============================================================================
// AGCN pipeline for MI455X (gfx1250, wave32, WMMA).
//  - All 7 GEMMs (~47 GFLOP) run through one bf16 WMMA kernel
//    (v_wmma_f32_16x16x32_bf16, fp32 accumulate), fused bias/ReLU epilogues.
//  - B operands are pre-packed TRANSPOSED (NxK) so both A and B stage into
//    LDS with contiguous b128 copies and fragments are contiguous b128 reads.
//  - 128x256 block tile, 64x64 wave tile -> 16 WMMA per wave per K-step.
//  - Final GEMM stores transposed (coalesced float4), fusing the output
//    transpose.
// ============================================================================

typedef unsigned short u16;
typedef __attribute__((ext_vector_type(16))) __bf16 bf16x16;
typedef __attribute__((ext_vector_type(8)))  float  f32x8;

union Frag8 { uint4 u; __bf16 h[8]; };

__device__ __forceinline__ u16 f32_to_bf16_bits(float f) {
  unsigned u = __float_as_uint(f);
  unsigned r = (u + 0x7FFFu + ((u >> 16) & 1u)) >> 16;  // round-to-nearest-even
  return (u16)r;
}

__device__ __forceinline__ bf16x16 make_frag(uint4 lo, uint4 hi) {
  Frag8 a, b;
  a.u = lo; b.u = hi;
  bf16x16 f;
#pragma unroll
  for (int i = 0; i < 8; ++i) { f[i] = a.h[i]; f[i + 8] = b.h[i]; }
  return f;
}

// ---------------------------------------------------------------------------
// GEMM: C(MxN, f32) = A(MxK bf16, row-major) * Bt(NxK bf16, row-major)^T
// Block tile 128(M) x 256(N), BK=32. 8 waves (2x4), each wave 64x64 =
// 16 WMMA tiles of 16x16.
// biasMode: 0 none, 1 bias[m], 2 bias[n].  doRelu: clamp at 0.
// storeT: write C[n*ldc + m] (transposed, coalesced float4) instead of
//         C[m*ldc + n].
// ---------------------------------------------------------------------------
#define BM 128
#define BN 256
#define BK 32
#define LDSS 40   // LDS row stride in halves (32 + 8 pad)

__global__ __launch_bounds__(256, 1) void gemm_bf16_wmma(
    const u16* __restrict__ A, int lda,
    const u16* __restrict__ Bt, int ldb,
    float* __restrict__ C, int ldc,
    int M, int N, int Kd,
    const float* __restrict__ bias, int biasMode, int doRelu, int storeT)
{
  __shared__ u16 As[BM * LDSS];   // As[m][k]
  __shared__ u16 Bs[BN * LDSS];   // Bs[n][k]

  const int tid   = threadIdx.x;
  const int lane  = tid & 31;
  const int wave  = tid >> 5;
  const int waveM = wave >> 2;        // 0..1
  const int waveN = wave & 3;         // 0..3
  const int lrow  = lane & 15;
  const int lhalf = lane >> 4;        // 0 or 1

  const int blockM = blockIdx.y * BM;
  const int blockN = blockIdx.x * BN;

  // global->LDS staging: A 128x32 (16 halves/thread), B 256x32 (32/thread)
  const int arow = tid >> 1;          // 0..127
  const int acol = (tid & 1) * 16;    // 0 or 16

  f32x8 acc[4][4] = {};

  const u16* pArow = A  + (size_t)(blockM + arow) * lda + acol;
  const u16* pBrow = Bt + (size_t)(blockN + tid) * ldb;

  for (int k0 = 0; k0 < Kd; k0 += BK) {
    const uint4* pa = (const uint4*)(pArow + k0);
    uint4 a0 = pa[0];
    uint4 a1 = pa[1];
    const uint4* pb = (const uint4*)(pBrow + k0);
    uint4 b0 = pb[0];
    uint4 b1 = pb[1];
    uint4 b2 = pb[2];
    uint4 b3 = pb[3];

    if (k0 + BK < Kd) {   // hint next K tile toward the caches
      __builtin_prefetch(pArow + k0 + BK, 0, 1);
      __builtin_prefetch(pBrow + k0 + BK, 0, 1);
    }

    __syncthreads();   // previous iteration's fragment reads complete

    *(uint4*)&As[arow * LDSS + acol]     = a0;
    *(uint4*)&As[arow * LDSS + acol + 8] = a1;
    {
      uint4* bs = (uint4*)&Bs[tid * LDSS];
      bs[0] = b0; bs[1] = b1; bs[2] = b2; bs[3] = b3;   // 0,8,16,24 halves
    }
    __syncthreads();

    // CDNA5 wave32 16-bit fragment layouts:
    //  A (16x32): lane%16 = M row; lane half selects K 0-7/16-23 vs 8-15/24-31
    //  B (32x16): lane%16 = N col; lane half selects K 0-7/16-23 vs 8-15/24-31
    bf16x16 afrag[4], bfrag[4];
#pragma unroll
    for (int ti = 0; ti < 4; ++ti) {
      const int am = waveM * 64 + ti * 16 + lrow;
      uint4 lo = *(const uint4*)&As[am * LDSS + lhalf * 8];
      uint4 hi = *(const uint4*)&As[am * LDSS + 16 + lhalf * 8];
      afrag[ti] = make_frag(lo, hi);
    }
#pragma unroll
    for (int tj = 0; tj < 4; ++tj) {
      const int bn = waveN * 64 + tj * 16 + lrow;
      uint4 lo = *(const uint4*)&Bs[bn * LDSS + lhalf * 8];
      uint4 hi = *(const uint4*)&Bs[bn * LDSS + 16 + lhalf * 8];
      bfrag[tj] = make_frag(lo, hi);
    }

#pragma unroll
    for (int ti = 0; ti < 4; ++ti)
#pragma unroll
      for (int tj = 0; tj < 4; ++tj)
        acc[ti][tj] = __builtin_amdgcn_wmma_f32_16x16x32_bf16(
            false, afrag[ti], false, bfrag[tj], (short)0, acc[ti][tj],
            false, false);
  }

  // Epilogue. C/D layout: VGPR r holds M = r + 8*lhalf, N = lane%16.
#pragma unroll
  for (int ti = 0; ti < 4; ++ti) {
    const int m0 = blockM + waveM * 64 + ti * 16 + lhalf * 8;
#pragma unroll
    for (int tj = 0; tj < 4; ++tj) {
      const int n = blockN + waveN * 64 + tj * 16 + lrow;
      const float bn = (biasMode == 2) ? bias[n] : 0.0f;
      float v[8];
#pragma unroll
      for (int r = 0; r < 8; ++r) {
        v[r] = acc[ti][tj][r] + bn;
        if (biasMode == 1) v[r] += bias[m0 + r];
        if (doRelu) v[r] = fmaxf(v[r], 0.0f);
      }
      if (storeT) {
        // lane's 8 values are consecutive in m -> two coalesced float4 stores
        float4 lo = make_float4(v[0], v[1], v[2], v[3]);
        float4 hi = make_float4(v[4], v[5], v[6], v[7]);
        float* p = C + (size_t)n * ldc + m0;
        *(float4*)p       = lo;
        *(float4*)(p + 4) = hi;
      } else {
#pragma unroll
        for (int r = 0; r < 8; ++r)
          C[(size_t)(m0 + r) * ldc + n] = v[r];
      }
    }
  }
}

// ---------------------------------------------------------------------------
// fp32 -> bf16 pack (plain)
// ---------------------------------------------------------------------------
__global__ void pack_bf16_k(const float* __restrict__ in, u16* __restrict__ out, long n) {
  long i = (long)blockIdx.x * blockDim.x + threadIdx.x;
  const long stride = (long)gridDim.x * blockDim.x;
  for (; i < n; i += stride) out[i] = f32_to_bf16_bits(in[i]);
}

// fp32 (RxC) -> bf16 (CxR), LDS-tiled transpose
__global__ __launch_bounds__(256) void packT_bf16_k(const float* __restrict__ in,
                                                    u16* __restrict__ out, int R, int C) {
  __shared__ float tile[32][33];
  const int r0 = blockIdx.y * 32, c0 = blockIdx.x * 32;
  const int tx = threadIdx.x, ty = threadIdx.y;   // 32 x 8
#pragma unroll
  for (int i = 0; i < 32; i += 8)
    tile[ty + i][tx] = in[(size_t)(r0 + ty + i) * C + (c0 + tx)];
  __syncthreads();
#pragma unroll
  for (int i = 0; i < 32; i += 8)
    out[(size_t)(c0 + ty + i) * R + (r0 + tx)] = f32_to_bf16_bits(tile[tx][ty + i]);
}

// ---------------------------------------------------------------------------
// Column softmax: P (RxC) fp32, softmax over rows, per column, in place.
// 256 threads = 64 columns x 4 row-chunks; coalesced along columns.
// ---------------------------------------------------------------------------
__global__ __launch_bounds__(256) void col_softmax_k(float* __restrict__ P, int R, int C) {
  __shared__ float redA[4][64];
  __shared__ float redB[4][64];
  const int lc = threadIdx.x & 63;
  const int rg = threadIdx.x >> 6;     // 0..3
  const int c  = blockIdx.x * 64 + lc;
  const int rows = R >> 2;
  const int r0 = rg * rows, r1 = r0 + rows;

  float mx = -3.4e38f;
  for (int r = r0; r < r1; ++r) mx = fmaxf(mx, P[(size_t)r * C + c]);
  redA[rg][lc] = mx;
  __syncthreads();
  const float gmax = fmaxf(fmaxf(redA[0][lc], redA[1][lc]),
                           fmaxf(redA[2][lc], redA[3][lc]));

  float s = 0.0f;
  for (int r = r0; r < r1; ++r) s += expf(P[(size_t)r * C + c] - gmax);
  redB[rg][lc] = s;
  __syncthreads();
  const float inv = 1.0f / (redB[0][lc] + redB[1][lc] + redB[2][lc] + redB[3][lc]);

  for (int r = r0; r < r1; ++r) {
    const size_t i = (size_t)r * C + c;
    P[i] = expf(P[i] - gmax) * inv;
  }
}

// Row softmax: one block per row, in place.
__global__ __launch_bounds__(256) void row_softmax_k(float* __restrict__ P, int C) {
  __shared__ float red[256];
  const int tid = threadIdx.x;
  float* row = P + (size_t)blockIdx.x * C;

  float mx = -3.4e38f;
  for (int c = tid; c < C; c += 256) mx = fmaxf(mx, row[c]);
  red[tid] = mx;
  __syncthreads();
  for (int off = 128; off > 0; off >>= 1) {
    if (tid < off) red[tid] = fmaxf(red[tid], red[tid + off]);
    __syncthreads();
  }
  const float gmax = red[0];
  __syncthreads();

  float s = 0.0f;
  for (int c = tid; c < C; c += 256) s += expf(row[c] - gmax);
  red[tid] = s;
  __syncthreads();
  for (int off = 128; off > 0; off >>= 1) {
    if (tid < off) red[tid] += red[tid + off];
    __syncthreads();
  }
  const float inv = 1.0f / red[0];
  __syncthreads();

  for (int c = tid; c < C; c += 256) row[c] = expf(row[c] - gmax) * inv;
}

// ===========================================================================
extern "C" void kernel_launch(void* const* d_in, const int* in_sizes, int n_in,
                              void* d_out, int out_size, void* d_ws, size_t ws_size,
                              hipStream_t stream) {
  const int CH = 1024;   // CHNN
  const int KK = 2048;   // K == L

  const float* vc    = (const float*)d_in[0];
  const float* vm    = (const float*)d_in[1];
  const float* W_ak  = (const float*)d_in[2];
  const float* b_ak  = (const float*)d_in[3];
  const float* W_c   = (const float*)d_in[4];
  const float* b_c   = (const float*)d_in[5];
  const float* W_mad = (const float*)d_in[6];
  const float* b_mad = (const float*)d_in[7];
  const float* W_gcn = (const float*)d_in[8];
  const float* b_gcn = (const float*)d_in[9];
  float* out = (float*)d_out;

  size_t off = 0;
  auto alloc = [&](size_t bytes) -> void* {
    void* p = (char*)d_ws + off;
    off += (bytes + 255) & ~(size_t)255;
    return p;
  };

  // ---- operand packs (A ops plain MxK; B ops transposed NxK) ----
  u16* Wak_bf    = (u16*)alloc((size_t)KK * CH * 2);   // A1: W_ak (2048x1024)
  u16* vcT_bf    = (u16*)alloc((size_t)KK * CH * 2);   // B1^T: vc^T (2048x1024)
  u16* Wc_bf     = (u16*)alloc((size_t)CH * CH * 2);   // A2
  u16* vmT_bf    = (u16*)alloc((size_t)KK * CH * 2);   // B2^T
  u16* Wmad_bf   = (u16*)alloc((size_t)CH * CH * 2);   // A4
  u16* WgcnT_bf  = (u16*)alloc((size_t)CH * CH * 2);   // B6^T
  // ---- fp32 intermediates + dependent packs ----
  float* vcw     = (float*)alloc((size_t)KK * KK * 4); // reused as adjacency A
  u16* ST_bf     = (u16*)alloc((size_t)KK * KK * 2);   // S^T; reused as softmax(A) pack
  float* vmr     = (float*)alloc((size_t)CH * KK * 4);
  u16* vmr_bf    = (u16*)alloc((size_t)CH * KK * 2);   // A3
  u16* vmrT_bf   = (u16*)alloc((size_t)KK * CH * 2);   // A6
  float* vma0    = (float*)alloc((size_t)CH * KK * 4);
  u16* vma0T_bf  = (u16*)alloc((size_t)KK * CH * 2);   // B4^T
  float* vma     = (float*)alloc((size_t)CH * KK * 4);
  u16* vmaT_bf   = (u16*)alloc((size_t)KK * CH * 2);   // A5 AND B5^T (shared!)
  float* x       = (float*)alloc((size_t)KK * CH * 4);
  u16* xT_bf     = (u16*)alloc((size_t)CH * KK * 2);   // B7^T
  (void)ws_size; (void)n_in; (void)in_sizes; (void)out_size;

  float* Aadj     = vcw;    // 2048x2048 fp32, reuse
  u16*   Asoft_bf = ST_bf;  // 2048x2048 bf16 (A7), reuse

  auto gemm = [&](const u16* A, int lda, const u16* Bt, int ldb, float* C, int ldc,
                  int M, int N, int Kd, const float* bias, int biasMode,
                  int relu, int storeT) {
    dim3 grid(N / BN, M / BM), block(256);
    gemm_bf16_wmma<<<grid, block, 0, stream>>>(A, lda, Bt, ldb, C, ldc, M, N, Kd,
                                               bias, biasMode, relu, storeT);
  };
  auto pack = [&](const float* in, u16* o, long n) {
    pack_bf16_k<<<1024, 256, 0, stream>>>(in, o, n);
  };
  auto packT = [&](const float* in, u16* o, int R, int C) {
    packT_bf16_k<<<dim3(C / 32, R / 32), dim3(32, 8), 0, stream>>>(in, o, R, C);
  };

  // ---- pack weights + input activations ----
  pack (W_ak,  Wak_bf,  (long)KK * CH);
  packT(vc,    vcT_bf,  CH, KK);
  pack (W_c,   Wc_bf,   (long)CH * CH);
  packT(vm,    vmT_bf,  CH, KK);
  pack (W_mad, Wmad_bf, (long)CH * CH);
  packT(W_gcn, WgcnT_bf, CH, CH);

  // 1) vcw = W_ak * vc + b_ak ; column softmax -> S ; pack S^T
  gemm(Wak_bf, CH, vcT_bf, CH, vcw, KK, KK, KK, CH, b_ak, 1, 0, 0);
  col_softmax_k<<<KK / 64, 256, 0, stream>>>(vcw, KK, KK);
  packT(vcw, ST_bf, KK, KK);

  // 2) vmr = relu(W_c * vm + b_c)
  gemm(Wc_bf, CH, vmT_bf, CH, vmr, KK, CH, KK, CH, b_c, 1, 1, 0);
  pack (vmr, vmr_bf, (long)CH * KK);
  packT(vmr, vmrT_bf, CH, KK);

  // 3) vma0 = vmr * S   (B^T = S^T)
  gemm(vmr_bf, KK, ST_bf, KK, vma0, KK, CH, KK, KK, nullptr, 0, 0, 0);
  packT(vma0, vma0T_bf, CH, KK);

  // 4) vma = W_mad * vma0 + b_mad
  gemm(Wmad_bf, CH, vma0T_bf, CH, vma, KK, CH, KK, CH, b_mad, 1, 0, 0);
  packT(vma, vmaT_bf, CH, KK);    // vma^T serves as A and B^T of GEMM 5

  // 5) A = vma^T * vma ; row softmax ; pack plain for GEMM 7
  gemm(vmaT_bf, CH, vmaT_bf, CH, Aadj, KK, KK, KK, CH, nullptr, 0, 0, 0);
  row_softmax_k<<<KK, 256, 0, stream>>>(Aadj, KK);
  pack(Aadj, Asoft_bf, (long)KK * KK);

  // 6) x = vmr^T * W_gcn + b_gcn (bias per column)
  gemm(vmrT_bf, CH, WgcnT_bf, CH, x, CH, KK, CH, CH, b_gcn, 2, 0, 0);
  packT(x, xT_bf, KK, CH);

  // 7) d_out = (softmax(A) * x)^T  -- transposed store fused into epilogue
  gemm(Asoft_bf, KK, xT_bf, KK, out, KK, KK, CH, KK, nullptr, 0, 0, 1);
}
